// Net_20993800143102
// MI455X (gfx1250) — matrix operand order repaired and checked by reading, not compiled, not run on hardware
//
#include <hip/hip_runtime.h>
#include <hip/hip_bf16.h>

typedef __attribute__((ext_vector_type(16))) _Float16 v16h;
typedef __attribute__((ext_vector_type(8)))  _Float16 v8h;
typedef __attribute__((ext_vector_type(8)))  float    v8f;

#define HCS   10
#define D_IN_ 128
#define G_    1024
#define NPG_  512
#define N_    (G_ * NPG_)        // 524288
#define E_    (N_ * 16)          // 8388608
#define KP1   461                // ceil(0.9*512)
#define KP2   415                // ceil(0.9*461)
#define KP3   374                // ceil(0.9*415)

#define AP    144                // padded halves per sA row (288 B, 72 dw: bank stride 8)
#define WP    144                // padded halves per sWt row

// ---------------------------------------------------------------------------
// Kernel 1: fused LayerNorm + dual GEMM  xn @ W1_rel, xn @ W1_root  via WMMA.
// Block = 256 threads = 8 waves; each wave computes a 16x16 (M x Npad) tile
// with two v_wmma_f32_16x16x32_f16 chains over K=128 (4 unrolled chunks).
// Fragments assembled with 16-byte LDS vector loads (layout-matched tiles).
// ---------------------------------------------------------------------------
__global__ __launch_bounds__(256) void k_ln_gemm(
    const float* __restrict__ x, const float* __restrict__ gamma,
    const float* __restrict__ beta, const float* __restrict__ Wrel,
    const float* __restrict__ Wroot, float* __restrict__ yrel,
    float* __restrict__ agg)
{
  __shared__ __align__(16) _Float16 sA[128][AP];      // normalized f16 rows (36 KB)
  __shared__ __align__(16) _Float16 sWt[2][16][WP];   // weights transposed [col][K] (9 KB)
  __shared__ float red[128][2][2];
  __shared__ float sMu[128], sRs[128];

  const int tid = threadIdx.x;
  const int blockRow0 = blockIdx.x * 128;

  // --- weights -> LDS transposed to [matrix][out-col(16)][K(128)] ---
  for (int idx = tid; idx < 2 * 16 * D_IN_; idx += 256) {
    int r = idx & 127;            // K row
    int c = (idx >> 7) & 15;      // out col (padded)
    int m = idx >> 11;            // which matrix
    float v = 0.0f;
    if (c < HCS) v = (m == 0 ? Wrel : Wroot)[r * HCS + c];
    sWt[m][c][r] = (_Float16)v;
  }

  // --- LayerNorm: 2 threads per row, 64 cols each ---
  const int row  = tid >> 1;
  const int half = tid & 1;
  const float* xr = x + (size_t)(blockRow0 + row) * D_IN_ + half * 64;
  float s = 0.0f, s2 = 0.0f;
  for (int c = 0; c < 64; ++c) { float v = xr[c]; s += v; s2 += v * v; }
  red[row][half][0] = s; red[row][half][1] = s2;
  __syncthreads();
  if (half == 0) {
    float ts  = red[row][0][0] + red[row][1][0];
    float ts2 = red[row][0][1] + red[row][1][1];
    float mu  = ts * (1.0f / D_IN_);
    float var = ts2 * (1.0f / D_IN_) - mu * mu;
    sMu[row] = mu;
    sRs[row] = rsqrtf(var + 1e-5f);
  }
  __syncthreads();
  {
    float mu = sMu[row], rs = sRs[row];
    for (int c = 0; c < 64; ++c) {
      int cc = half * 64 + c;
      float v = (xr[c] - mu) * rs * gamma[cc] + beta[cc];
      sA[row][cc] = (_Float16)v;
    }
  }
  __syncthreads();

  // --- WMMA: wave w handles rows [w*16, w*16+16) ---
  const int wave = tid >> 5;
  const int lane = tid & 31;
  const int m    = lane & 15;          // A row within tile / B col / C col
  const int hi   = lane >> 4;          // lane-half select
  const int arow = wave * 16 + m;

  // A fragment: lanes<16: v0-3 = K0..7, v4-7 = K16..23; hi lanes +8.
  // Both runs contiguous -> two 16B LDS loads.
  auto loadA = [&](int kc) -> v16h {
    v8h a0 = *(const v8h*)&sA[arow][kc * 32 + hi * 8];
    v8h a1 = *(const v8h*)&sA[arow][kc * 32 + 16 + hi * 8];
    return __builtin_shufflevector(a0, a1, 0, 1, 2, 3, 4, 5, 6, 7,
                                   8, 9, 10, 11, 12, 13, 14, 15);
  };
  // B fragment: same K order as A (lanes<16: K0..7, K16..23; hi lanes +8)
  // -> two 16B LDS loads.
  auto loadB = [&](int mtx, int kc) -> v16h {
    v8h b0 = *(const v8h*)&sWt[mtx][m][kc * 32 + hi * 8];
    v8h b1 = *(const v8h*)&sWt[mtx][m][kc * 32 + 16 + hi * 8];
    return __builtin_shufflevector(b0, b1, 0, 1, 2, 3, 4, 5, 6, 7,
                                   8, 9, 10, 11, 12, 13, 14, 15);
  };

  v8f crel = {}; v8f croot = {};
#pragma unroll
  for (int kc = 0; kc < 4; ++kc) {
    v16h a  = loadA(kc);
    v16h br = loadB(0, kc);
    v16h bo = loadB(1, kc);
    crel  = __builtin_amdgcn_wmma_f32_16x16x32_f16(false, a, false, br,
                                                   (short)0, crel,  false, false);
    croot = __builtin_amdgcn_wmma_f32_16x16x32_f16(false, a, false, bo,
                                                   (short)0, croot, false, false);
  }

  // --- C layout: N = lane&15, VGPR j -> M = j + hi*8 ---
  if (m < HCS) {
#pragma unroll
    for (int j = 0; j < 8; ++j) {
      int orow = blockRow0 + wave * 16 + j + hi * 8;
      yrel[(size_t)orow * HCS + m] = crel[j];
      agg [(size_t)orow * HCS + m] = croot[j];   // agg initialized with root term
    }
  }
}

// ---------------------------------------------------------------------------
// Edge scatter: agg[d] += w_e * yrel[s], with optional 1- or 2-level id remap.
// ---------------------------------------------------------------------------
__global__ void k_edge(const int* __restrict__ src, const int* __restrict__ dst,
                       const float* __restrict__ w, const int* __restrict__ map1,
                       const int* __restrict__ map2, const float* __restrict__ yrel,
                       float* __restrict__ agg, int ne)
{
  int e = blockIdx.x * blockDim.x + threadIdx.x;
  if (e >= ne) return;
  int s = src[e], d = dst[e];
  if (map1) { s = map1[s]; if (s < 0) return; d = map1[d]; if (d < 0) return; }
  if (map2) { s = map2[s]; if (s < 0) return; d = map2[d]; if (d < 0) return; }
  float wv = w[e];
  const float* ys = yrel + (size_t)s * HCS;
  float*       ad = agg  + (size_t)d * HCS;
#pragma unroll
  for (int c = 0; c < HCS; ++c) atomicAdd(&ad[c], ys[c] * wv);
}

// Per-node 10x10 projections for levels 2/3: yrel = x@Wrel, agg = x@Wroot.
__global__ void k_ymat(const float* __restrict__ xin, int nn,
                       const float* __restrict__ Wrel, const float* __restrict__ Wroot,
                       float* __restrict__ yrel, float* __restrict__ agg)
{
  int i = blockIdx.x * blockDim.x + threadIdx.x;
  if (i >= nn) return;
  float xi[HCS];
#pragma unroll
  for (int j = 0; j < HCS; ++j) xi[j] = xin[(size_t)i * HCS + j];
#pragma unroll
  for (int c = 0; c < HCS; ++c) {
    float sr = 0.0f, so = 0.0f;
#pragma unroll
    for (int j = 0; j < HCS; ++j) {
      sr += xi[j] * Wrel [j * HCS + c];
      so += xi[j] * Wroot[j * HCS + c];
    }
    yrel[(size_t)i * HCS + c] = sr;
    agg [(size_t)i * HCS + c] = so;
  }
}

// agg = relu(agg + bias)
__global__ void k_combine(float* __restrict__ agg, const float* __restrict__ b, int total)
{
  int i = blockIdx.x * blockDim.x + threadIdx.x;
  if (i >= total) return;
  float v = agg[i] + b[i % HCS];
  agg[i] = v > 0.0f ? v : 0.0f;
}

// ---------------------------------------------------------------------------
// TopK pool: per-graph scores, O(n^2) rank (== new local id per top_k+perm
// semantics), gate kept nodes by tanh score, emit id remap (-1 = dropped).
// ---------------------------------------------------------------------------
__global__ __launch_bounds__(256) void k_pool(
    const float* __restrict__ xin, int n, int k, const float* __restrict__ pw,
    float* __restrict__ xout, int* __restrict__ idxout)
{
  __shared__ float sc[NPG_];
  int g = blockIdx.x;
  float pwl[HCS]; float nrm = 0.0f;
#pragma unroll
  for (int c = 0; c < HCS; ++c) { pwl[c] = pw[c]; nrm += pwl[c] * pwl[c]; }
  float rn = 1.0f / sqrtf(nrm);

  for (int i = threadIdx.x; i < n; i += blockDim.x) {
    const float* xi = xin + (size_t)(g * n + i) * HCS;
    float d = 0.0f;
#pragma unroll
    for (int c = 0; c < HCS; ++c) d += xi[c] * pwl[c];
    sc[i] = tanhf(d * rn);
  }
  __syncthreads();

  for (int i = threadIdx.x; i < n; i += blockDim.x) {
    float si = sc[i];
    int rank = 0;
    for (int j = 0; j < n; ++j) {
      float sj = sc[j];
      rank += (sj > si) || (sj == si && j < i);   // top_k index tie-break
    }
    if (rank < k) {
      const float* xi = xin  + (size_t)(g * n + i)    * HCS;
      float*       xo = xout + (size_t)(g * k + rank) * HCS;
#pragma unroll
      for (int c = 0; c < HCS; ++c) xo[c] = xi[c] * si;
      idxout[g * n + i] = g * k + rank;
    } else {
      idxout[g * n + i] = -1;
    }
  }
}

// Readout: rsum[g, 0:10] += max_i x, rsum[g, 10:20] += mean_i x
__global__ __launch_bounds__(256) void k_readout(const float* __restrict__ xp, int k,
                                                 float* __restrict__ rsum)
{
  __shared__ float pmax[16][16];
  __shared__ float psum[16][16];
  int g   = blockIdx.x;
  int c   = threadIdx.x & 15;
  int grp = threadIdx.x >> 4;
  float mx = -3.4e38f, sm = 0.0f;
  if (c < HCS) {
    for (int i = grp; i < k; i += 16) {
      float v = xp[(size_t)(g * k + i) * HCS + c];
      mx = v > mx ? v : mx;
      sm += v;
    }
  }
  pmax[grp][c] = mx; psum[grp][c] = sm;
  __syncthreads();
  if (grp == 0 && c < HCS) {
    float tmx = pmax[0][c], tsm = psum[0][c];
#pragma unroll
    for (int r = 1; r < 16; ++r) { tmx = fmaxf(tmx, pmax[r][c]); tsm += psum[r][c]; }
    rsum[g * 20 + c]        += tmx;
    rsum[g * 20 + 10 + c]   += tsm / (float)k;
  }
}

__global__ void k_zero(float* p, int n)
{
  int i = blockIdx.x * blockDim.x + threadIdx.x;
  if (i < n) p[i] = 0.0f;
}

// MLP head per graph: relu(sum) -> 20->80 -> 80->320 -> 320->10
__global__ __launch_bounds__(256) void k_mlp(
    const float* __restrict__ rsum,
    const float* __restrict__ W1, const float* __restrict__ b1,
    const float* __restrict__ W2, const float* __restrict__ b2,
    const float* __restrict__ W3, const float* __restrict__ b3,
    float* __restrict__ out)
{
  __shared__ float h0[20], h1[80], h2[320];
  int g = blockIdx.x, t = threadIdx.x;
  if (t < 20) { float v = rsum[g * 20 + t]; h0[t] = v > 0.0f ? v : 0.0f; }
  __syncthreads();
  if (t < 80) {
    float a = b1[t];
    for (int j = 0; j < 20; ++j) a += h0[j] * W1[j * 80 + t];
    h1[t] = a > 0.0f ? a : 0.0f;
  }
  __syncthreads();
  for (int o = t; o < 320; o += 256) {
    float a = b2[o];
    for (int j = 0; j < 80; ++j) a += h1[j] * W2[j * 320 + o];
    h2[o] = a > 0.0f ? a : 0.0f;
  }
  __syncthreads();
  if (t < HCS) {
    float a = b3[t];
    for (int j = 0; j < 320; ++j) a += h2[j] * W3[j * HCS + t];
    out[g * HCS + t] = a;
  }
}

// ---------------------------------------------------------------------------
extern "C" void kernel_launch(void* const* d_in, const int* in_sizes, int n_in,
                              void* d_out, int out_size, void* d_ws, size_t ws_size,
                              hipStream_t stream)
{
  (void)in_sizes; (void)n_in; (void)out_size; (void)ws_size;
  const float* x      = (const float*)d_in[0];
  const int*   ei     = (const int*)  d_in[1];
  const float* ea     = (const float*)d_in[2];
  const float* gamma  = (const float*)d_in[4];
  const float* beta   = (const float*)d_in[5];
  const float* W1rel  = (const float*)d_in[6];
  const float* b1rel  = (const float*)d_in[7];
  const float* W1root = (const float*)d_in[8];
  const float* W2rel  = (const float*)d_in[9];
  const float* b2rel  = (const float*)d_in[10];
  const float* W2root = (const float*)d_in[11];
  const float* W3rel  = (const float*)d_in[12];
  const float* b3rel  = (const float*)d_in[13];
  const float* W3root = (const float*)d_in[14];
  const float* p1     = (const float*)d_in[15];
  const float* p2     = (const float*)d_in[16];
  const float* p3     = (const float*)d_in[17];
  const float* L1W    = (const float*)d_in[18];
  const float* L1b    = (const float*)d_in[19];
  const float* L2W    = (const float*)d_in[20];
  const float* L2b    = (const float*)d_in[21];
  const float* L3W    = (const float*)d_in[22];
  const float* L3b    = (const float*)d_in[23];
  float* out = (float*)d_out;

  char* ws = (char*)d_ws;
  size_t off = 0;
  auto alloc = [&](size_t bytes) -> void* {
    void* p = ws + off;
    off = (off + bytes + 255) & ~(size_t)255;
    return p;
  };
  float* yrel  = (float*)alloc((size_t)N_ * HCS * sizeof(float));
  float* agg   = (float*)alloc((size_t)N_ * HCS * sizeof(float));
  float* xpool = (float*)alloc((size_t)N_ * HCS * sizeof(float));
  int*   idx1  = (int*)  alloc((size_t)N_ * sizeof(int));
  int*   idx2  = (int*)  alloc((size_t)G_ * KP1 * sizeof(int));
  int*   idx3  = (int*)  alloc((size_t)G_ * KP2 * sizeof(int));
  float* rsum  = (float*)alloc((size_t)G_ * 20 * sizeof(float));

  const int* srcp = ei;
  const int* dstp = ei + E_;

  k_zero<<<(G_ * 20 + 255) / 256, 256, 0, stream>>>(rsum, G_ * 20);

  // ---- level 1: LN + WMMA projections, scatter, combine, pool, readout ----
  k_ln_gemm<<<N_ / 128, 256, 0, stream>>>(x, gamma, beta, W1rel, W1root, yrel, agg);
  k_edge<<<E_ / 256, 256, 0, stream>>>(srcp, dstp, ea, nullptr, nullptr, yrel, agg, E_);
  k_combine<<<(N_ * HCS + 255) / 256, 256, 0, stream>>>(agg, b1rel, N_ * HCS);
  k_pool<<<G_, 256, 0, stream>>>(agg, NPG_, KP1, p1, xpool, idx1);
  k_readout<<<G_, 256, 0, stream>>>(xpool, KP1, rsum);

  // ---- level 2 ----
  int n1 = G_ * KP1;
  k_ymat<<<(n1 + 255) / 256, 256, 0, stream>>>(xpool, n1, W2rel, W2root, yrel, agg);
  k_edge<<<E_ / 256, 256, 0, stream>>>(srcp, dstp, ea, idx1, nullptr, yrel, agg, E_);
  k_combine<<<(n1 * HCS + 255) / 256, 256, 0, stream>>>(agg, b2rel, n1 * HCS);
  k_pool<<<G_, 256, 0, stream>>>(agg, KP1, KP2, p2, xpool, idx2);
  k_readout<<<G_, 256, 0, stream>>>(xpool, KP2, rsum);

  // ---- level 3 ----
  int n2 = G_ * KP2;
  k_ymat<<<(n2 + 255) / 256, 256, 0, stream>>>(xpool, n2, W3rel, W3root, yrel, agg);
  k_edge<<<E_ / 256, 256, 0, stream>>>(srcp, dstp, ea, idx1, idx2, yrel, agg, E_);
  k_combine<<<(n2 * HCS + 255) / 256, 256, 0, stream>>>(agg, b3rel, n2 * HCS);
  k_pool<<<G_, 256, 0, stream>>>(agg, KP2, KP3, p3, xpool, idx3);
  k_readout<<<G_, 256, 0, stream>>>(xpool, KP3, rsum);

  // ---- MLP head ----
  k_mlp<<<G_, 256, 0, stream>>>(rsum, L1W, L1b, L2W, L2b, L3W, L3b, out);
}
